// INRLoe_30966714204325
// MI455X (gfx1250) — matrix operand order repaired and checked by reading, not compiled, not run on hardware
//
#include <hip/hip_runtime.h>

typedef _Float16 v16h  __attribute__((ext_vector_type(16)));
typedef float    v8f   __attribute__((ext_vector_type(8)));
typedef unsigned int u32x4 __attribute__((ext_vector_type(4)));
typedef float    f32x4 __attribute__((ext_vector_type(4)));
typedef _Float16 h4    __attribute__((ext_vector_type(4)));

#define NS 64        // samples
#define NP 8192      // points per sample
#define KTOT 328     // sum of top-k counts over 5 layers
#define BPITCH 1028  // bias row pitch (256*4 + 3, padded)

struct GatePtrs { const float* gw[5]; const float* gb[5]; };
struct BiasPtrs { const float* bs[5]; };

// ---------------------------------------------------------------------------
// Gate GEMV + top-k (bitonic sort by |g| in LDS). One block per (n, layer).
// ---------------------------------------------------------------------------
__global__ __launch_bounds__(256) void k_gate(const float* __restrict__ lat,
                                              GatePtrs gp,
                                              float* __restrict__ tkval,
                                              int* __restrict__ tkidx) {
  const int Es[5] = {8, 16, 64, 256, 1024};
  const int Ks[5] = {4, 4, 32, 32, 256};
  const int Ko[5] = {0, 4, 8, 40, 72};
  const int b = blockIdx.x;
  const int n = b / 5, l = b % 5;
  const int E = Es[l], K = Ks[l], koff = Ko[l];
  const int tid = threadIdx.x;

  __shared__ float sKey[1024];
  __shared__ float sVal[1024];
  __shared__ int   sIdx[1024];
  __shared__ float sLat[64];

  if (tid < 64) sLat[tid] = lat[(n * 5 + l) * 64 + tid];
  __syncthreads();

  const float* gw = gp.gw[l];
  const float* gb = gp.gb[l];
  for (int e = tid; e < E; e += 256) {
    const float* wr = gw + e * 64;
    float a = gb[e];
#pragma unroll 8
    for (int i = 0; i < 64; ++i) a += sLat[i] * wr[i];
    sVal[e] = a; sKey[e] = fabsf(a); sIdx[e] = e;
  }
  __syncthreads();

  // bitonic sort descending by key (E is a power of two)
  for (int size = 2; size <= E; size <<= 1) {
    for (int stride = size >> 1; stride > 0; stride >>= 1) {
      for (int i = tid; i < (E >> 1); i += 256) {
        int pos = 2 * i - (i & (stride - 1));
        int oth = pos + stride;
        bool desc = ((pos & size) == 0);
        float kp = sKey[pos], kq = sKey[oth];
        if ((kp < kq) == desc) {
          sKey[pos] = kq; sKey[oth] = kp;
          float vp = sVal[pos]; sVal[pos] = sVal[oth]; sVal[oth] = vp;
          int ip = sIdx[pos]; sIdx[pos] = sIdx[oth]; sIdx[oth] = ip;
        }
      }
      __syncthreads();
    }
  }
  for (int j = tid; j < K; j += 256) {
    tkval[n * KTOT + koff + j] = sVal[j];
    tkidx[n * KTOT + koff + j] = sIdx[j];
  }
}

// ---------------------------------------------------------------------------
// Sparse expert blend: Wc[n][m] = sum_j val_j * wbank[idx_j*M + m]
// dst16 != nullptr -> emit f16 (WMMA B operand), else f32.
// ---------------------------------------------------------------------------
__global__ __launch_bounds__(256) void k_combine(const float* __restrict__ wbank,
                                                 const float* __restrict__ tkval,
                                                 const int* __restrict__ tkidx,
                                                 int koff, int K, int M,
                                                 float* __restrict__ dst32,
                                                 _Float16* __restrict__ dst16) {
  const int n = blockIdx.y;
  const int tid = threadIdx.x;
  __shared__ float sv[256];
  __shared__ int   si[256];
  if (tid < K) {
    sv[tid] = tkval[n * KTOT + koff + tid];
    si[tid] = tkidx[n * KTOT + koff + tid];
  }
  __syncthreads();
  const int M4 = M >> 2;
  const int m4 = blockIdx.x * 256 + tid;
  if (m4 >= M4) return;
  const f32x4* wb = (const f32x4*)wbank;
  f32x4 acc = {0.f, 0.f, 0.f, 0.f};
  for (int j = 0; j < K; ++j) {
    float v = sv[j];
    f32x4 wv = wb[(size_t)si[j] * M4 + m4];
    acc += wv * v;
  }
  if (dst16) {
    h4 o;
    o[0] = (_Float16)acc[0]; o[1] = (_Float16)acc[1];
    o[2] = (_Float16)acc[2]; o[3] = (_Float16)acc[3];
    *(h4*)(dst16 + (size_t)n * M + 4 * m4) = o;
  } else {
    ((f32x4*)dst32)[(size_t)n * M4 + m4] = acc;
  }
}

// ---------------------------------------------------------------------------
// Blend biases for all layers. One block per sample.
// ---------------------------------------------------------------------------
__global__ __launch_bounds__(256) void k_bias(BiasPtrs bp,
                                              const float* __restrict__ tkval,
                                              const int* __restrict__ tkidx,
                                              float* __restrict__ Bc) {
  const int Ks[5] = {4, 4, 32, 32, 256};
  const int Ko[5] = {0, 4, 8, 40, 72};
  const int Bo[5] = {0, 256, 512, 768, 1024};
  const int n = blockIdx.x, tid = threadIdx.x;
  for (int l = 0; l < 5; ++l) {
    const int dout = (l < 4) ? 256 : 3;
    const float* bs = bp.bs[l];
    const int K = Ks[l], koff = Ko[l];
    for (int o = tid; o < dout; o += 256) {
      float a = 0.f;
      for (int j = 0; j < K; ++j) {
        int e = tkidx[n * KTOT + koff + j];
        a += tkval[n * KTOT + koff + j] * bs[e * dout + o];
      }
      Bc[n * BPITCH + Bo[l] + o] = a;
    }
  }
}

// ---------------------------------------------------------------------------
// Fused 5-layer forward. Block = 64 points of one sample, 8 waves.
// Layers 1-3: WMMA f32<-f16 16x16x32, W staged in LDS in two K-halves.
// B fragments are explicitly double-buffered so ds_loads for fragment ct+1
// are in flight behind the WMMA consuming fragment ct.
// ---------------------------------------------------------------------------
__global__ __launch_bounds__(256) void k_main(const float* __restrict__ coords,
                                              const float* __restrict__ Wc0,
                                              const float* __restrict__ Wc4,
                                              const _Float16* __restrict__ Wc16,
                                              const float* __restrict__ Bc,
                                              float* __restrict__ out) {
  // pitches padded so full-wave b128 LDS ops hit the 2-cycle bank minimum
  __shared__ __attribute__((aligned(16))) _Float16 sX[64 * 264];   // 33 KB
  __shared__ __attribute__((aligned(16))) _Float16 sW[256 * 136];  // 68 KB
  __shared__ float sC[128];

  const int n = blockIdx.y;
  const int rowBase = blockIdx.x * 64;
  const int tid = threadIdx.x;
  const int lane = tid & 31;
  const int w = tid >> 5;
  const int lane15 = lane & 15;
  const int halfSel = lane >> 4;
  const int r0 = (w & 3) * 16;        // wave's 16 output rows
  const int colHalf = (w >> 2) * 128; // wave's 128 output cols

  if (tid < 128) sC[tid] = coords[((size_t)n * NP + rowBase) * 2 + tid];

  // warm L1/L2 for the first W half while layer 0 runs
  {
    const _Float16* p0 = Wc16 + (size_t)n * 65536;
    __builtin_prefetch(p0 + (size_t)tid * 128, 0, 3);        // 256 x 256B
    __builtin_prefetch(p0 + (size_t)tid * 128 + 64, 0, 3);
  }
  __syncthreads();

  // ---- layer 0: [2 -> 256], VALU + sin, write f16 activations to LDS ----
  {
    const int o = tid;
    const float w0 = Wc0[n * 512 + 2 * o];
    const float w1 = Wc0[n * 512 + 2 * o + 1];
    const float b  = Bc[n * BPITCH + o];
    for (int r = 0; r < 64; ++r) {
      float v = __sinf(30.f * (sC[2 * r] * w0 + sC[2 * r + 1] * w1 + b));
      sX[r * 264 + o] = (_Float16)v;
    }
  }

  // ---- layers 1..3: [256 -> 256] WMMA ----
  for (int L = 1; L <= 3; ++L) {
    const _Float16* WcL = Wc16 + ((size_t)(L - 1) * NS + n) * 65536;
    v8f acc[8];
#pragma unroll
    for (int t = 0; t < 8; ++t)
      acc[t] = (v8f){0.f, 0.f, 0.f, 0.f, 0.f, 0.f, 0.f, 0.f};

    for (int kh = 0; kh < 2; ++kh) {
      __syncthreads();  // previous consumers of sW / producers of sX done
      // cooperative stage: 256 rows x 128 f16 of W into LDS (coalesced b128)
#pragma unroll
      for (int i = 0; i < 16; ++i) {
        int u = i * 256 + tid;
        int row = u >> 4, col = u & 15;
        u32x4 d = *(const u32x4*)(WcL + (size_t)row * 256 + kh * 128 + col * 8);
        *(u32x4*)(sW + row * 136 + col * 8) = d;
      }
      __syncthreads();

      // prefetch the next W chunk (other K-half, or next layer's first half)
      {
        const _Float16* nxt = nullptr;
        if (kh == 0)      nxt = WcL + 128;                           // same layer, half 1
        else if (L < 3)   nxt = Wc16 + ((size_t)L * NS + n) * 65536; // next layer, half 0
        if (nxt) {
          __builtin_prefetch(nxt + (size_t)tid * 128, 0, 3);
          __builtin_prefetch(nxt + (size_t)tid * 128 + 64, 0, 3);
        }
      }

      for (int k0 = 0; k0 < 128; k0 += 32) {
        union Frag { v16h h; u32x4 u[2]; };
        // A fragment: lanes 0-15 rows M, K 0-15; lanes 16-31 K 16-31
        Frag A;
        const _Float16* ap = sX + (r0 + lane15) * 264 + kh * 128 + k0 + halfSel * 16;
        A.u[0] = *(const u32x4*)(ap);
        A.u[1] = *(const u32x4*)(ap + 8);

        // B fragments: lane holds 16 consecutive K of column (colHalf+ct*16+lane15)
        const _Float16* bbase = sW + (colHalf + lane15) * 136 + k0 + halfSel * 16;
        Frag B0, B1;
        B0.u[0] = *(const u32x4*)(bbase);
        B0.u[1] = *(const u32x4*)(bbase + 8);
#pragma unroll
        for (int ct = 0; ct < 8; ++ct) {
          if (ct + 1 < 8) {   // issue loads for fragment ct+1 before consuming ct
            const _Float16* nb = bbase + (ct + 1) * (16 * 136);
            if (ct & 1) {
              B0.u[0] = *(const u32x4*)(nb);
              B0.u[1] = *(const u32x4*)(nb + 8);
            } else {
              B1.u[0] = *(const u32x4*)(nb);
              B1.u[1] = *(const u32x4*)(nb + 8);
            }
          }
          const v16h bh = (ct & 1) ? B1.h : B0.h;
          acc[ct] = __builtin_amdgcn_wmma_f32_16x16x32_f16(
              false, A.h, false, bh, (short)0, acc[ct], false, false);
        }
      }
    }

    __syncthreads();  // all waves finished reading sX for this layer
#pragma unroll
    for (int ct = 0; ct < 8; ++ct) {
      const int c = colHalf + ct * 16 + lane15;
      const float bv = Bc[n * BPITCH + 256 * L + c];
      const int rb = r0 + halfSel * 8;
#pragma unroll
      for (int j = 0; j < 8; ++j) {
        float x = __sinf(30.f * (acc[ct][j] + bv));
        sX[(rb + j) * 264 + c] = (_Float16)x;
      }
    }
  }
  __syncthreads();

  // ---- layer 4: [256 -> 3] head, fp32 output ----
  if (tid < 64) {
    const int r = tid;
    const float* W4 = Wc4 + n * 768;
    float a0 = Bc[n * BPITCH + 1024 + 0];
    float a1 = Bc[n * BPITCH + 1024 + 1];
    float a2 = Bc[n * BPITCH + 1024 + 2];
    for (int i = 0; i < 256; i += 8) {
      union { u32x4 u; _Float16 h[8]; } V;
      V.u = *(const u32x4*)(sX + r * 264 + i);
#pragma unroll
      for (int q = 0; q < 8; ++q) {
        float x = (float)V.h[q];
        a0 += x * W4[i + q];
        a1 += x * W4[256 + i + q];
        a2 += x * W4[512 + i + q];
      }
    }
    size_t ob = ((size_t)n * NP + rowBase + r) * 3;
    out[ob + 0] = a0; out[ob + 1] = a1; out[ob + 2] = a2;
  }
}

// ---------------------------------------------------------------------------
extern "C" void kernel_launch(void* const* d_in, const int* in_sizes, int n_in,
                              void* d_out, int out_size, void* d_ws, size_t ws_size,
                              hipStream_t stream) {
  (void)in_sizes; (void)n_in; (void)out_size; (void)ws_size;
  const float* latents = (const float*)d_in[0];
  const float* coords  = (const float*)d_in[1];
  GatePtrs gp; BiasPtrs bp; const float* wbank[5];
  for (int l = 0; l < 5; ++l) {
    gp.gw[l]  = (const float*)d_in[2 + 2 * l];
    gp.gb[l]  = (const float*)d_in[3 + 2 * l];
    wbank[l]  = (const float*)d_in[12 + 2 * l];
    bp.bs[l]  = (const float*)d_in[13 + 2 * l];
  }

  // scratch layout (bytes, all 256-aligned)
  char* ws = (char*)d_ws;
  float*     tkval = (float*)(ws + 0);        // 64*328*4   = 83968
  int*       tkidx = (int*)  (ws + 83968);    // 83968
  float*     Bc    = (float*)(ws + 167936);   // 64*1028*4  = 263168
  float*     Wc0   = (float*)(ws + 431104);   // 64*512*4   = 131072
  float*     Wc4   = (float*)(ws + 562176);   // 64*768*4   = 196608
  _Float16*  Wc16  = (_Float16*)(ws + 758784);// 3*64*65536*2 = 24 MB

  const int Ks[5] = {4, 4, 32, 32, 256};
  const int Ko[5] = {0, 4, 8, 40, 72};

  k_gate<<<NS * 5, 256, 0, stream>>>(latents, gp, tkval, tkidx);

  // layer 0 weights (M=512, f32)
  k_combine<<<dim3(1, NS), 256, 0, stream>>>(wbank[0], tkval, tkidx,
                                             Ko[0], Ks[0], 512, Wc0, nullptr);
  // layers 1..3 weights (M=65536, f16 for WMMA)
  for (int l = 1; l <= 3; ++l)
    k_combine<<<dim3(64, NS), 256, 0, stream>>>(wbank[l], tkval, tkidx,
                                                Ko[l], Ks[l], 65536, nullptr,
                                                Wc16 + (size_t)(l - 1) * NS * 65536);
  // layer 4 weights (M=768, f32)
  k_combine<<<dim3(1, NS), 256, 0, stream>>>(wbank[4], tkval, tkidx,
                                             Ko[4], Ks[4], 768, Wc4, nullptr);

  k_bias<<<NS, 256, 0, stream>>>(bp, tkval, tkidx, Bc);

  k_main<<<dim3(NP / 64, NS), 256, 0, stream>>>(coords, Wc0, Wc4, Wc16, Bc,
                                                (float*)d_out);
}
